// MSDeformableAttention_84782654423159
// MI455X (gfx1250) — compile-verified, matrix operand order checked by reading, not run on hardware
//
#include <hip/hip_runtime.h>
#include <stdint.h>

// ---------------- problem constants (from reference) ----------------
#define BS     4
#define EMBED  256
#define HEADS  8
#define LEVELS 4
#define POINTS 4
#define HD     32          // EMBED / HEADS
#define LQ     12240       // sum of level H*W
#define MROWS  (BS * LQ)   // 48960, GEMM M dimension
#define NCAT   384         // fused off(256) + attn(128) output width

__constant__ int kLH[LEVELS]     = {96, 48, 24, 12};
__constant__ int kLW[LEVELS]     = {96, 48, 24, 12};
__constant__ int kLSTART[LEVELS] = {0, 9216, 11520, 12096};

// ---------------- bf16 helpers (raw-bit form, no __bf16 scalar ops) ----------------
typedef __bf16 v16bf __attribute__((ext_vector_type(16)));
typedef float  v8f   __attribute__((ext_vector_type(8)));

union Frag16 { v16bf v; uint32_t u[8]; };

__device__ __forceinline__ unsigned short f2bf_bits(float f) {
    union { float f; uint32_t u; } a; a.f = f;
    uint32_t r = a.u + 0x7FFFu + ((a.u >> 16) & 1u);   // round-to-nearest-even
    return (unsigned short)(r >> 16);
}
__device__ __forceinline__ float bf2f(unsigned short b) {
    union { uint32_t u; float f; } a; a.u = ((uint32_t)b) << 16;
    return a.f;
}

// 4 contiguous A elements -> 2 dwords of packed bf16 bits
__device__ __forceinline__ uint2 pack4_bf(const float* __restrict__ p) {
    float4 f = *(const float4*)p;                       // global_load_b128
    uint2 r;
    r.x = (uint32_t)f2bf_bits(f.x) | ((uint32_t)f2bf_bits(f.y) << 16);
    r.y = (uint32_t)f2bf_bits(f.z) | ((uint32_t)f2bf_bits(f.w) << 16);
    return r;
}
__device__ __forceinline__ uint2 pack4_bf(const unsigned short* __restrict__ p) {
    return *(const uint2*)p;                            // already bf16 bits (b64 load)
}

// ---------------- WMMA GEMM: C[M,N] = A[M,K] @ B[K,N] + bias[N] ----------------
// Block tile 64(M) x 128(N), 128 threads = 4 waves; wave = 16-row strip x 128 cols.
// Per 32-wide K panel: 1 A fragment reused across 8 v_wmma_f32_16x16x32_bf16.
#define LDA 40   // LDS halfword row stride: 80B = 16B-aligned, 16-row bank-conflict-free

template <typename AT, bool OUT_BF16>
__global__ __launch_bounds__(128)
void wmma_gemm_kernel(const AT* __restrict__ A, const float* __restrict__ B,
                      const float* __restrict__ bias, void* __restrict__ Cout,
                      int M, int N, int K) {
    __shared__ unsigned short lA[64 * LDA];    // [m][k] bf16 bits
    __shared__ unsigned short lBT[128 * LDA];  // [n][k] bf16 bits (B transposed)

    const int tid  = threadIdx.x;
    const int lane = tid & 31;
    const int wave = tid >> 5;            // 0..3
    const int tM   = blockIdx.y * 64;
    const int tN   = blockIdx.x * 128;
    const int mrow = wave * 16;

    v8f z = {0.f,0.f,0.f,0.f,0.f,0.f,0.f,0.f};
    v8f acc[8] = {z, z, z, z, z, z, z, z};

    for (int kk = 0; kk < K; kk += 32) {
        // stage A: 64 rows x 8 chunks-of-4 = 512 chunks, 4 per thread (exact)
        #pragma unroll
        for (int r = 0; r < 4; ++r) {
            int i  = tid + r * 128;
            int m  = i >> 3;
            int k4 = (i & 7) * 4;
            uint2 d = pack4_bf(&A[(size_t)(tM + m) * K + kk + k4]);
            *(uint2*)&lA[m * LDA + k4] = d;            // ds_store_b64
        }
        // stage B transposed: 32 k-rows x 32 chunks-of-4-n = 1024 chunks, 8/thread
        #pragma unroll
        for (int r = 0; r < 8; ++r) {
            int i  = tid + r * 128;
            int k  = i >> 5;
            int n4 = (i & 31) * 4;
            float4 f = *(const float4*)&B[(size_t)(kk + k) * N + tN + n4];
            lBT[(n4 + 0) * LDA + k] = f2bf_bits(f.x);
            lBT[(n4 + 1) * LDA + k] = f2bf_bits(f.y);
            lBT[(n4 + 2) * LDA + k] = f2bf_bits(f.z);
            lBT[(n4 + 3) * LDA + k] = f2bf_bits(f.w);
        }
        // speculative prefetch of next K panel of A (global_prefetch_b8)
        if (kk + 32 < K)
            __builtin_prefetch(&A[(size_t)(tM + (tid & 63)) * K + kk + 32], 0, 1);
        __syncthreads();

        // ISA 16-bit fragment layout: VGPR j holds
        //   K = (j>=4?16:0) + (lane>=16?8:0) + (j&3)*2 + {0,1}   (pairs contiguous)
        const int kb = (lane >> 4) * 8;
        Frag16 afr;
        {
            const uint32_t* base = (const uint32_t*)&lA[(mrow + (lane & 15)) * LDA];
            #pragma unroll
            for (int j = 0; j < 8; ++j) {
                int kOff = ((j >= 4) ? 16 : 0) + kb + (j & 3) * 2;
                afr.u[j] = base[kOff >> 1];
            }
        }
        #pragma unroll
        for (int t = 0; t < 8; ++t) {
            Frag16 bfr;
            const uint32_t* base =
                (const uint32_t*)&lBT[(t * 16 + (lane & 15)) * LDA];
            #pragma unroll
            for (int j = 0; j < 8; ++j) {
                int kOff = ((j >= 4) ? 16 : 0) + kb + (j & 3) * 2;
                bfr.u[j] = base[kOff >> 1];
            }
            acc[t] = __builtin_amdgcn_wmma_f32_16x16x32_bf16(
                false, afr.v, false, bfr.v, (short)0, acc[t], false, false);
        }
        __syncthreads();
    }

    // C/D layout: vgpr r, lane l -> m = r + (l>=16 ? 8 : 0), n = l & 15
    #pragma unroll
    for (int t = 0; t < 8; ++t) {
        int n  = tN + t * 16 + (lane & 15);
        float bv = bias[n];
        #pragma unroll
        for (int r = 0; r < 8; ++r) {
            int m = tM + mrow + r + ((lane >> 4) ? 8 : 0);
            float val = acc[t][r] + bv;
            if (OUT_BF16)
                ((unsigned short*)Cout)[(size_t)m * N + n] = f2bf_bits(val);
            else
                ((float*)Cout)[(size_t)m * N + n] = val;
        }
    }
}

// ---------------- weight concat: Wcat[256][384] = [W_off | W_attn] ----------------
__global__ __launch_bounds__(256)
void concat_weights_kernel(const float* __restrict__ W_off, const float* __restrict__ b_off,
                           const float* __restrict__ W_attn, const float* __restrict__ b_attn,
                           float* __restrict__ Wcat, float* __restrict__ bcat) {
    int i = blockIdx.x * 256 + threadIdx.x;
    if (i < 256 * NCAT) {
        int k = i / NCAT, n = i - k * NCAT;
        Wcat[i] = (n < 256) ? W_off[k * 256 + n] : W_attn[k * 128 + (n - 256)];
    }
    if (i < NCAT) bcat[i] = (i < 256) ? b_off[i] : b_attn[i - 256];
}

// ---------------- deformable sampling: one wave32 per (b,q,head), lane=channel ----------------
__device__ __forceinline__ float fetch_v(const unsigned short* __restrict__ v,
                                         int b, int start, int x, int y,
                                         int Wi, int Hi, int ch) {
    if (x < 0 || y < 0 || x >= Wi || y >= Hi) return 0.f;
    size_t idx = ((size_t)(b * LQ + start + y * Wi + x)) * EMBED + ch;
    return bf2f(v[idx]);
}

__global__ __launch_bounds__(256)
void msda_sample_kernel(const float* __restrict__ refpts,        // [BS,LQ,L,2]
                        const unsigned short* __restrict__ v,    // bf16 [BS*LQ,256]
                        const float* __restrict__ offattn,       // [BS*LQ,384]
                        unsigned short* __restrict__ acc)        // bf16 [BS*LQ,256]
{
    const int item = blockIdx.x * 8 + (threadIdx.x >> 5);  // (b*LQ+q)*HEADS + h
    const int lane = threadIdx.x & 31;                     // channel within head
    const int bq   = item >> 3;
    const int h    = item & 7;
    const int b    = bq / LQ;
    const int ch   = h * HD + lane;

    const float* row  = offattn + (size_t)bq * NCAT;
    const float* orow = row + h * 32;          // offsets (l,p,2)
    const float* arow = row + 256 + h * 16;    // attn logits (l*p)

    // softmax over the 16 (level,point) logits for this head
    float logit[16];
    float mx = -1e30f;
    #pragma unroll
    for (int i = 0; i < 16; ++i) { logit[i] = arow[i]; mx = fmaxf(mx, logit[i]); }
    float s = 0.f;
    #pragma unroll
    for (int i = 0; i < 16; ++i) { logit[i] = __expf(logit[i] - mx); s += logit[i]; }
    const float inv = 1.f / s;

    const float* rrow = refpts + (size_t)bq * (LEVELS * 2);

    float a = 0.f;
    #pragma unroll
    for (int l = 0; l < LEVELS; ++l) {
        const int Wi = kLW[l], Hi = kLH[l], start = kLSTART[l];
        const float Wf = (float)Wi, Hf = (float)Hi;
        const float rx = rrow[l * 2 + 0], ry = rrow[l * 2 + 1];
        #pragma unroll
        for (int p = 0; p < POINTS; ++p) {
            const int ip = l * POINTS + p;
            const float lx = rx + orow[ip * 2 + 0] / Wf;
            const float ly = ry + orow[ip * 2 + 1] / Hf;
            // grid_sample(align_corners=False): gx = loc*W - 0.5
            const float gx = lx * Wf - 0.5f;
            const float gy = ly * Hf - 0.5f;
            const float fx = floorf(gx), fy = floorf(gy);
            const float wx = gx - fx, wy = gy - fy;
            const int x0 = (int)fx, y0 = (int)fy;
            const float w00 = (1.f - wx) * (1.f - wy);
            const float w10 = wx * (1.f - wy);
            const float w01 = (1.f - wx) * wy;
            const float w11 = wx * wy;
            float smp = fetch_v(v, b, start, x0,     y0,     Wi, Hi, ch) * w00
                      + fetch_v(v, b, start, x0 + 1, y0,     Wi, Hi, ch) * w10
                      + fetch_v(v, b, start, x0,     y0 + 1, Wi, Hi, ch) * w01
                      + fetch_v(v, b, start, x0 + 1, y0 + 1, Wi, Hi, ch) * w11;
            a += (logit[ip] * inv) * smp;
        }
    }
    acc[(size_t)bq * EMBED + ch] = f2bf_bits(a);
}

// ---------------- launcher ----------------
extern "C" void kernel_launch(void* const* d_in, const int* in_sizes, int n_in,
                              void* d_out, int out_size, void* d_ws, size_t ws_size,
                              hipStream_t stream) {
    const float* query  = (const float*)d_in[0];
    const float* refpts = (const float*)d_in[1];
    const float* value  = (const float*)d_in[2];
    // d_in[3] = value_spatial_shapes: compile-time constants, ignored
    const float* W_off  = (const float*)d_in[4];
    const float* b_off  = (const float*)d_in[5];
    const float* W_attn = (const float*)d_in[6];
    const float* b_attn = (const float*)d_in[7];
    const float* W_val  = (const float*)d_in[8];
    const float* b_val  = (const float*)d_in[9];
    const float* W_out  = (const float*)d_in[10];
    const float* b_out  = (const float*)d_in[11];

    // workspace layout (bytes):
    //   Wcat    f32 [256,384]    :    393,216   @ 0
    //   bcat    f32 [384]        :      1,536   @ 393,216
    //   v       bf16 [48960,256] : 25,067,520   @ 394,752
    //   offattn f32  [48960,384] : 75,202,560   @ 25,462,272
    //   acc     bf16 [48960,256] : 25,067,520   @ 100,664,832   (total ~125.7 MB)
    char* ws = (char*)d_ws;
    float*          Wcat_ws    = (float*)(ws);
    float*          bcat_ws    = (float*)(ws + 393216);
    unsigned short* v_ws       = (unsigned short*)(ws + 394752);
    float*          offattn_ws = (float*)(ws + 25462272);
    unsigned short* acc_ws     = (unsigned short*)(ws + 100664832);

    const dim3 blk(128);
    const int  mb = MROWS / 64;  // 765

    // build fused [W_off | W_attn] weight + bias
    concat_weights_kernel<<<dim3((256 * NCAT) / 256), dim3(256), 0, stream>>>(
        W_off, b_off, W_attn, b_attn, Wcat_ws, bcat_ws);

    // v = value @ W_val + b_val  -> bf16 intermediate
    wmma_gemm_kernel<float, true><<<dim3(EMBED / 128, mb), blk, 0, stream>>>(
        value, W_val, b_val, (void*)v_ws, MROWS, EMBED, EMBED);

    // [off | attn] = query @ Wcat + bcat  -> f32 (single pass over query)
    wmma_gemm_kernel<float, false><<<dim3(NCAT / 128, mb), blk, 0, stream>>>(
        query, Wcat_ws, bcat_ws, (void*)offattn_ws, MROWS, NCAT, EMBED);

    // deformable bilinear sampling + softmax-weighted accumulation -> bf16 acc
    msda_sample_kernel<<<dim3((MROWS * HEADS) / 8), dim3(256), 0, stream>>>(
        refpts, v_ws, offattn_ws, acc_ws);

    // out = acc @ W_out + b_out  -> f32 output
    wmma_gemm_kernel<unsigned short, false><<<dim3(EMBED / 128, mb), blk, 0, stream>>>(
        acc_ws, W_out, b_out, d_out, MROWS, EMBED, EMBED);
}
